// DBGNN_16724602650672
// MI455X (gfx1250) — matrix operand order repaired and checked by reading, not compiled, not run on hardware
//
#include <hip/hip_runtime.h>
#include <cstdint>

#define DEV __device__ __forceinline__

typedef __attribute__((ext_vector_type(16))) __bf16 v16bf;
typedef __attribute__((ext_vector_type(8)))  __bf16 v8bf;
typedef __attribute__((ext_vector_type(8)))  float  v8f;
typedef __attribute__((ext_vector_type(4)))  float  f32x4;

DEV __bf16 f2bf(float f) {
  unsigned u = __builtin_bit_cast(unsigned, f);
  unsigned r = (u + 0x7FFFu + ((u >> 16) & 1u)) >> 16;
  unsigned short s = (unsigned short)r;
  return __builtin_bit_cast(__bf16, s);
}
DEV float bf2f(__bf16 b) {
  unsigned short s = __builtin_bit_cast(unsigned short, b);
  unsigned u = ((unsigned)s) << 16;
  return __builtin_bit_cast(float, u);
}

// A-fragment (16x32, M=lane&15) from an fp32 row, scaled (mean = msg * icnt).
// kbase = kstep*32 + (lane>>4)*8 ; element e: K = kbase + (e>>3)*16 + (e&7)
DEV v16bf load_a_f32(const float* __restrict__ rowp, float scale, int kbase) {
  v16bf f;
#pragma unroll
  for (int h = 0; h < 2; ++h) {
    const f32x4* p = (const f32x4*)(rowp + kbase + h * 16);
    f32x4 lo = p[0], hi = p[1];
    f[h * 8 + 0] = f2bf(lo.x * scale); f[h * 8 + 1] = f2bf(lo.y * scale);
    f[h * 8 + 2] = f2bf(lo.z * scale); f[h * 8 + 3] = f2bf(lo.w * scale);
    f[h * 8 + 4] = f2bf(hi.x * scale); f[h * 8 + 5] = f2bf(hi.y * scale);
    f[h * 8 + 6] = f2bf(hi.z * scale); f[h * 8 + 7] = f2bf(hi.w * scale);
  }
  return f;
}

// A-fragment from a bf16 row (root features), same K mapping.
DEV v16bf load_a_bf16(const __bf16* __restrict__ rowp, int kbase) {
  v16bf f;
#pragma unroll
  for (int h = 0; h < 2; ++h) {
    v8bf v = *(const v8bf*)(rowp + kbase + h * 16);
#pragma unroll
    for (int i = 0; i < 8; ++i) f[h * 8 + i] = v[i];
  }
  return f;
}

// ---------------- embed: h[n, f*16+d] = x[n,f]*W[f,d] + b[f,d]  (bf16 out) ----
__global__ void embed_kernel(const float* __restrict__ x, const float* __restrict__ Wc,
                             const float* __restrict__ bc, __bf16* __restrict__ h, int n) {
  size_t idx = (size_t)blockIdx.x * blockDim.x + threadIdx.x;
  if (idx >= (size_t)n * 128) return;
  int node = (int)(idx >> 7);
  int j = (int)(idx & 127);
  int f = j >> 4, d = j & 15;
  float v = x[(size_t)node * 8 + f] * Wc[f * 16 + d] + bc[f * 16 + d];
  h[idx] = f2bf(v);
}

// ---------------- edge scatter: msg[dst] += h_src[src] (fp32 atomics) --------
__global__ void scatter_kernel(const __bf16* __restrict__ hsrc, const int* __restrict__ src,
                               const int* __restrict__ dst, float* __restrict__ msg) {
  int e = blockIdx.x;
  int t = threadIdx.x; // 128
  int s = src[e], d = dst[e];
  atomicAdd(&msg[(size_t)d * 128 + t], bf2f(hsrc[(size_t)s * 128 + t]));
}

__global__ void count_kernel(const int* __restrict__ dst, float* __restrict__ cnt, int E) {
  int e = blockIdx.x * blockDim.x + threadIdx.x;
  if (e < E) atomicAdd(&cnt[dst[e]], 1.0f);
}

__global__ void invert_kernel(float* __restrict__ cnt, int n) {
  int i = blockIdx.x * blockDim.x + threadIdx.x;
  if (i < n) cnt[i] = 1.0f / fmaxf(cnt[i], 1.0f);
}

// ---------------- weight pack into B-fragment layout -------------------------
// dst element idx = ((tile*4+ks)*32+lane)*16+e ; N = tile*16+(lane&15),
// K = ks*32 + (lane>>4)*8 + (e>>3)*16 + (e&7) ;  v = (A[K,n] (+B[K,n])) * scale
__global__ void pack_w_kernel(const float* __restrict__ A, const float* __restrict__ B,
                              float scale, __bf16* __restrict__ dst,
                              int ntiles, int ldn, int ncols) {
  int idx = blockIdx.x * blockDim.x + threadIdx.x;
  int total = ntiles * 4 * 32 * 16;
  if (idx >= total) return;
  int e = idx & 15;
  int lane = (idx >> 4) & 31;
  int ks = (idx >> 9) & 3;
  int tile = idx >> 11;
  int n = tile * 16 + (lane & 15);
  int K = ks * 32 + ((lane >> 4) * 8) + ((e >> 3) * 16) + (e & 7);
  float v = 0.0f;
  if (n < ncols) {
    v = A[(size_t)K * ldn + n];
    if (B) v += B[(size_t)K * ldn + n];
    v *= scale;
  }
  dst[idx] = f2bf(v);
}

__global__ void bias_t_kernel(const float* __restrict__ b_lin, float* __restrict__ bt) {
  int i = blockIdx.x * blockDim.x + threadIdx.x;
  if (i >= 2 * 128) return;
  int l = i >> 7, j = i & 127;
  bt[i] = 0.5f * (b_lin[(l * 4 + 0) * 128 + j] + b_lin[(l * 4 + 2) * 128 + j]);
}

// ---------------- fused SAGE GEMM: out = meanA@WA + root@WR (+ meanB@WB) + b --
// One wave -> 16 rows x 128 cols strip. K=128 via 4 x wmma_f32_16x16x32_bf16.
// Packed weights are staged in LDS once per block (8 waves share them), so the
// WMMA stream is fed by ds_load_b128 instead of per-wave L2 round-trips.
// PRECONDITION: nrows % 16 == 0 (holds for NC/NT/NP) -> no per-lane guards,
// EXEC stays all-1 everywhere (WMMA requirement) and epilogue is branch-free.
template <bool HASB>
__global__ __launch_bounds__(256) void sage_gemm_kernel(
    const float* __restrict__ msgA, const float* __restrict__ icntA, const uint4* __restrict__ WA,
    const __bf16* __restrict__ rootX, const uint4* __restrict__ WR,
    const float* __restrict__ msgB, const float* __restrict__ icntB, const uint4* __restrict__ WB,
    const float* __restrict__ bias, __bf16* __restrict__ out, int nrows) {
  extern __shared__ __bf16 smem[]; // (HASB?3:2) * 32KB
  // ---- cooperative weight staging (all threads; before any early return) ----
  {
    uint4* s4 = (uint4*)smem;
#pragma unroll 1
    for (int i = threadIdx.x; i < 2048; i += 256) { // 2048 uint4 = 32KB / matrix
      s4[i] = WA[i];
      s4[2048 + i] = WR[i];
      if (HASB) s4[4096 + i] = WB[i];
    }
  }
  __syncthreads();

  const int wave = threadIdx.x >> 5;
  const int lane = threadIdx.x & 31;
  const int row_base = (blockIdx.x * 8 + wave) * 16;
  if (row_base >= nrows) return; // wave-uniform: EXEC stays all-1 for live waves
  const int mloc = lane & 15;
  const int m = row_base + mloc; // always < nrows (nrows % 16 == 0)
  const int khalf = (lane >> 4) * 8;

  const v16bf* LWA = (const v16bf*)smem; // 1024 fragments per matrix
  const v16bf* LWR = LWA + 1024;
  const v16bf* LWB = LWA + 2048;

  v16bf aM[4], aR[4], aB[4];
  {
    const float scA = icntA[m];
    const float scB = HASB ? icntB[m] : 0.0f;
    const float* rowA = msgA + (size_t)m * 128;
    const __bf16* rowR = rootX + (size_t)m * 128;
    const float* rowB = HASB ? (msgB + (size_t)m * 128) : nullptr;
#pragma unroll
    for (int ks = 0; ks < 4; ++ks) {
      int kb = ks * 32 + khalf;
      aM[ks] = load_a_f32(rowA, scA, kb);
      aR[ks] = load_a_bf16(rowR, kb);
      if (HASB) aB[ks] = load_a_f32(rowB, scB, kb);
    }
  }

#pragma unroll
  for (int nt = 0; nt < 8; ++nt) {
    v8f acc = {0.f, 0.f, 0.f, 0.f, 0.f, 0.f, 0.f, 0.f};
#pragma unroll
    for (int ks = 0; ks < 4; ++ks) {
      int widx = (nt * 4 + ks) * 32 + lane;
      acc = __builtin_amdgcn_wmma_f32_16x16x32_bf16(false, aM[ks], false, LWA[widx],
                                                    (short)0, acc, false, false);
      acc = __builtin_amdgcn_wmma_f32_16x16x32_bf16(false, aR[ks], false, LWR[widx],
                                                    (short)0, acc, false, false);
      if (HASB)
        acc = __builtin_amdgcn_wmma_f32_16x16x32_bf16(false, aB[ks], false, LWB[widx],
                                                      (short)0, acc, false, false);
    }
    int col = nt * 16 + mloc;
    float bc = bias[col];
#pragma unroll
    for (int r = 0; r < 8; ++r) {
      int row = row_base + r + (lane >> 4) * 8;
      out[(size_t)row * 128 + col] = f2bf(acc[r] + bc);
    }
  }
}

// ---------------- head GEMM: logits[NT,16] = h_t @ Wout_pk + b_out -----------
__global__ __launch_bounds__(256) void out_gemm_kernel(
    const __bf16* __restrict__ ht, const uint4* __restrict__ Wout,
    const float* __restrict__ bout, float* __restrict__ logits, int nrows) {
  extern __shared__ __bf16 smem[]; // 4KB
  {
    uint4* s4 = (uint4*)smem;
    if (threadIdx.x < 256) { // 256 uint4 = 4KB
      s4[threadIdx.x] = Wout[threadIdx.x];
    }
  }
  __syncthreads();

  const int wave = threadIdx.x >> 5;
  const int lane = threadIdx.x & 31;
  const int row_base = (blockIdx.x * 8 + wave) * 16;
  if (row_base >= nrows) return;
  const int mloc = lane & 15;
  const int m = row_base + mloc; // always < nrows (nrows % 16 == 0)
  const int khalf = (lane >> 4) * 8;
  const v16bf* LW = (const v16bf*)smem;

  v16bf a[4];
  const __bf16* rowR = ht + (size_t)m * 128;
#pragma unroll
  for (int ks = 0; ks < 4; ++ks)
    a[ks] = load_a_bf16(rowR, ks * 32 + khalf);

  v8f acc = {0.f, 0.f, 0.f, 0.f, 0.f, 0.f, 0.f, 0.f};
#pragma unroll
  for (int ks = 0; ks < 4; ++ks)
    acc = __builtin_amdgcn_wmma_f32_16x16x32_bf16(false, a[ks], false, LW[ks * 32 + lane],
                                                  (short)0, acc, false, false);
  int col = mloc;
  // cols >= 10 are never consumed by softmax; clamp the index so the bias load
  // stays unguarded (full EXEC, no divergent branch).
  float bc = bout[col < 10 ? col : 9];
#pragma unroll
  for (int r = 0; r < 8; ++r) {
    int row = row_base + r + (lane >> 4) * 8;
    logits[(size_t)row * 16 + col] = acc[r] + bc;
  }
}

// ---------------- row softmax over 10 logits ---------------------------------
__global__ void softmax_kernel(const float* __restrict__ logits, float* __restrict__ out, int nrows) {
  int i = blockIdx.x * blockDim.x + threadIdx.x;
  if (i >= nrows) return;
  const float* l = logits + (size_t)i * 16;
  float mx = l[0];
#pragma unroll
  for (int j = 1; j < 10; ++j) mx = fmaxf(mx, l[j]);
  float e[10], s = 0.0f;
#pragma unroll
  for (int j = 0; j < 10; ++j) { e[j] = __expf(l[j] - mx); s += e[j]; }
  float inv = 1.0f / s;
#pragma unroll
  for (int j = 0; j < 10; ++j) out[(size_t)i * 10 + j] = e[j] * inv;
}

extern "C" void kernel_launch(void* const* d_in, const int* in_sizes, int n_in,
                              void* d_out, int out_size, void* d_ws, size_t ws_size,
                              hipStream_t stream) {
  (void)in_sizes; (void)n_in; (void)out_size;
  constexpr int NC = 100000, NT = 300000, NP = 50000, D = 128;
  constexpr int EM = 300000, EI = 600000;

  const float* x_c   = (const float*)d_in[0];
  const float* x_t   = (const float*)d_in[1];
  const float* x_p   = (const float*)d_in[2];
  const float* W_col = (const float*)d_in[3];
  const float* b_col = (const float*)d_in[4];
  const float* Wn    = (const float*)d_in[5];
  const float* Wr    = (const float*)d_in[6];
  const float* b_lin = (const float*)d_in[7];
  const float* W_out = (const float*)d_in[8];
  const float* b_out = (const float*)d_in[9];
  const int* em_src  = (const int*)d_in[10];
  const int* em_dst  = (const int*)d_in[11];
  const int* ei_src  = (const int*)d_in[12];
  const int* ei_dst  = (const int*)d_in[13];
  float* outp = (float*)d_out;

  char* base = (char*)d_ws;
  size_t off = 0;
  auto alloc = [&](size_t bytes) -> void* {
    void* p = base + off;
    off = (off + bytes + 255) & ~(size_t)255;
    return p;
  };

  __bf16* hc[2] = {(__bf16*)alloc((size_t)NC * D * 2), (__bf16*)alloc((size_t)NC * D * 2)};
  __bf16* ht[2] = {(__bf16*)alloc((size_t)NT * D * 2), (__bf16*)alloc((size_t)NT * D * 2)};
  __bf16* hp[2] = {(__bf16*)alloc((size_t)NP * D * 2), (__bf16*)alloc((size_t)NP * D * 2)};
  float* msg_t0 = (float*)alloc((size_t)NT * D * 4);
  float* msg_t2 = (float*)alloc((size_t)NT * D * 4);
  float* msg_c  = (float*)alloc((size_t)NC * D * 4);
  float* msg_p  = (float*)alloc((size_t)NP * D * 4);
  float* icnt_t0 = (float*)alloc((size_t)NT * 4);
  float* icnt_t2 = (float*)alloc((size_t)NT * 4);
  float* icnt_c  = (float*)alloc((size_t)NC * 4);
  float* icnt_p  = (float*)alloc((size_t)NP * 4);
  __bf16* wpk    = (__bf16*)alloc((size_t)15 * 16384 * 2); // 14 D×D matrices + head
  float* bt      = (float*)alloc((size_t)2 * D * 4);
  float* logits  = (float*)alloc((size_t)NT * 16 * 4);
  if (off > ws_size) return; // workspace too small; bail rather than corrupt

  auto slot = [&](int s) -> uint4* { return (uint4*)(wpk + (size_t)s * 16384); };
  const size_t MW = (size_t)D * D; // 16384 floats per weight matrix

  // ---- pack weights (bf16 B-fragment layout); HeteroConv 0.5 folded in -----
  for (int l = 0; l < 2; ++l) {
    int s0 = l * 7;
    pack_w_kernel<<<64, 256, 0, stream>>>(Wn + (l * 4 + 1) * MW, nullptr, 1.0f, (__bf16*)slot(s0 + 0), 8, D, D);
    pack_w_kernel<<<64, 256, 0, stream>>>(Wr + (l * 4 + 1) * MW, nullptr, 1.0f, (__bf16*)slot(s0 + 1), 8, D, D);
    pack_w_kernel<<<64, 256, 0, stream>>>(Wn + (l * 4 + 3) * MW, nullptr, 1.0f, (__bf16*)slot(s0 + 2), 8, D, D);
    pack_w_kernel<<<64, 256, 0, stream>>>(Wr + (l * 4 + 3) * MW, nullptr, 1.0f, (__bf16*)slot(s0 + 3), 8, D, D);
    pack_w_kernel<<<64, 256, 0, stream>>>(Wn + (l * 4 + 0) * MW, nullptr, 0.5f, (__bf16*)slot(s0 + 4), 8, D, D);
    pack_w_kernel<<<64, 256, 0, stream>>>(Wn + (l * 4 + 2) * MW, nullptr, 0.5f, (__bf16*)slot(s0 + 5), 8, D, D);
    pack_w_kernel<<<64, 256, 0, stream>>>(Wr + (l * 4 + 0) * MW, Wr + (l * 4 + 2) * MW, 0.5f,
                                          (__bf16*)slot(s0 + 6), 8, D, D);
  }
  pack_w_kernel<<<8, 256, 0, stream>>>(W_out, nullptr, 1.0f, (__bf16*)slot(14), 1, 10, 10);
  bias_t_kernel<<<1, 256, 0, stream>>>(b_lin, bt);

  // ---- degree counts (topology-only, once per call) -------------------------
  hipMemsetAsync(icnt_t0, 0, (size_t)NT * 4, stream);
  hipMemsetAsync(icnt_t2, 0, (size_t)NT * 4, stream);
  hipMemsetAsync(icnt_c, 0, (size_t)NC * 4, stream);
  hipMemsetAsync(icnt_p, 0, (size_t)NP * 4, stream);
  count_kernel<<<(EM + 255) / 256, 256, 0, stream>>>(em_dst, icnt_t0, EM);
  count_kernel<<<(EM + 255) / 256, 256, 0, stream>>>(em_src, icnt_c, EM);
  count_kernel<<<(EI + 255) / 256, 256, 0, stream>>>(ei_dst, icnt_t2, EI);
  count_kernel<<<(EI + 255) / 256, 256, 0, stream>>>(ei_src, icnt_p, EI);
  invert_kernel<<<(NT + 255) / 256, 256, 0, stream>>>(icnt_t0, NT);
  invert_kernel<<<(NT + 255) / 256, 256, 0, stream>>>(icnt_t2, NT);
  invert_kernel<<<(NC + 255) / 256, 256, 0, stream>>>(icnt_c, NC);
  invert_kernel<<<(NP + 255) / 256, 256, 0, stream>>>(icnt_p, NP);

  // ---- column embeddings ----------------------------------------------------
  embed_kernel<<<(int)(((size_t)NC * D + 255) / 256), 256, 0, stream>>>(x_c, W_col + 0 * 128, b_col + 0 * 128, hc[0], NC);
  embed_kernel<<<(int)(((size_t)NT * D + 255) / 256), 256, 0, stream>>>(x_t, W_col + 1 * 128, b_col + 1 * 128, ht[0], NT);
  embed_kernel<<<(int)(((size_t)NP * D + 255) / 256), 256, 0, stream>>>(x_p, W_col + 2 * 128, b_col + 2 * 128, hp[0], NP);

  // ---- 2 hetero-SAGE layers -------------------------------------------------
  int cur = 0;
  for (int l = 0; l < 2; ++l) {
    int nxt = cur ^ 1;
    hipMemsetAsync(msg_t0, 0, (size_t)NT * D * 4, stream);
    hipMemsetAsync(msg_t2, 0, (size_t)NT * D * 4, stream);
    hipMemsetAsync(msg_c, 0, (size_t)NC * D * 4, stream);
    hipMemsetAsync(msg_p, 0, (size_t)NP * D * 4, stream);

    scatter_kernel<<<EM, 128, 0, stream>>>(hc[cur], em_src, em_dst, msg_t0); // c -> t
    scatter_kernel<<<EM, 128, 0, stream>>>(ht[cur], em_dst, em_src, msg_c);  // t -> c
    scatter_kernel<<<EI, 128, 0, stream>>>(hp[cur], ei_src, ei_dst, msg_t2); // p -> t
    scatter_kernel<<<EI, 128, 0, stream>>>(ht[cur], ei_dst, ei_src, msg_p);  // t -> p

    sage_gemm_kernel<false><<<(NC + 127) / 128, 256, 2 * 32768, stream>>>(
        msg_c, icnt_c, slot(l * 7 + 0), hc[cur], slot(l * 7 + 1),
        nullptr, nullptr, nullptr, b_lin + (l * 4 + 1) * D, hc[nxt], NC);
    sage_gemm_kernel<false><<<(NP + 127) / 128, 256, 2 * 32768, stream>>>(
        msg_p, icnt_p, slot(l * 7 + 2), hp[cur], slot(l * 7 + 3),
        nullptr, nullptr, nullptr, b_lin + (l * 4 + 3) * D, hp[nxt], NP);
    sage_gemm_kernel<true><<<(NT + 127) / 128, 256, 3 * 32768, stream>>>(
        msg_t0, icnt_t0, slot(l * 7 + 4), ht[cur], slot(l * 7 + 6),
        msg_t2, icnt_t2, slot(l * 7 + 5), bt + l * D, ht[nxt], NT);
    cur = nxt;
  }

  // ---- head + softmax -------------------------------------------------------
  out_gemm_kernel<<<(NT + 127) / 128, 256, 4096, stream>>>(ht[cur], slot(14), b_out, logits, NT);
  softmax_kernel<<<(NT + 255) / 256, 256, 0, stream>>>(logits, outp, NT);
}
